// StudentGRU_82712480186641
// MI455X (gfx1250) — compile-verified, matrix operand order checked by reading, not run on hardware
//
#include <hip/hip_runtime.h>

// ---------------- problem constants ----------------
#define BATCH   4096
#define HDIM    512
#define G3      1536        // 3*H
#define IN_DIM  4
#define OUT_DIM 2
#define T_OBS   30
#define T_PRED  60

// ---------------- tiling ----------------
#define TPB        512      // 16 wave32 per block
#define WAVES      16
#define TILE_B     64       // batch rows per block
#define MT_TILES   4        // TILE_B/16
#define MPAIRS     2        // process 2 M-tiles per unit (B-fragment reuse)
#define JGROUPS    32       // HDIM/16 column groups
#define UNITS_PW   4        // MPAIRS*JGROUPS/WAVES
#define KT         16       // HDIM/32 k-tiles
#define LDS_STRIDE 520      // bf16 elems per h row (512 + pad, mult of 8)

#define FRAG_MAT   786432   // (G3/16)*(HDIM/32)*32*16 bf16 elems per packed matrix
#define SMEM_BYTES (4*TILE_B*LDS_STRIDE*2 + TILE_B*OUT_DIM*4)

typedef __bf16 bf16;
typedef __bf16 bf16x8  __attribute__((ext_vector_type(8)));
typedef __bf16 bf16x16 __attribute__((ext_vector_type(16)));
typedef float  f32x8   __attribute__((ext_vector_type(8)));

// ---------------- helpers ----------------
__device__ __forceinline__ f32x8 wmma_bf16(bf16x16 a, bf16x16 b, f32x8 c) {
    // D = A(16x32) * B(32x16) + C, fp32 accumulate
    return __builtin_amdgcn_wmma_f32_16x16x32_bf16(
        /*neg_a=*/false, a, /*neg_b=*/false, b,
        /*c_mod=*/(short)0, c, /*reuse_a=*/false, /*reuse_b=*/false);
}

__device__ __forceinline__ bf16x16 combine(bf16x8 lo, bf16x8 hi) {
    bf16x16 r;
#pragma unroll
    for (int i = 0; i < 8; ++i) { r[i] = lo[i]; r[i + 8] = hi[i]; }
    return r;
}

// A fragment (16x32, bf16) from LDS-resident h tile.
// Lane L<16: M=L,    kh=0 ; Lane L>=16: M=L-16, kh=8 (ISA 16-bit A striping)
__device__ __forceinline__ bf16x16 ldsA(const bf16* h, int mt, int kt, int lane) {
    int row = mt * 16 + (lane & 15);
    int kh  = (lane >> 4) << 3;
    const bf16* p = h + row * LDS_STRIDE + kt * 32 + kh;
    bf16x8 lo = *(const bf16x8*)p;
    bf16x8 hi = *(const bf16x8*)(p + 16);
    return combine(lo, hi);
}

// B fragment (32x16) from pre-packed weights: 32 lanes x 16 bf16, contiguous.
__device__ __forceinline__ bf16x16 ldB(const bf16* frags, int nt, int kt, int lane) {
    const bf16* p = frags + ((size_t)((nt * KT + kt) * 32 + lane)) * 16;
    bf16x8 lo = *(const bf16x8*)p;
    bf16x8 hi = *(const bf16x8*)(p + 8);
    return combine(lo, hi);
}

__device__ __forceinline__ float sigmoidf_(float x) {
    return 1.0f / (1.0f + __expf(-x));
}
__device__ __forceinline__ float tanhf_(float x) {
    float ax = fabsf(x);
    float e  = __expf(-2.0f * ax);
    float t  = (1.0f - e) / (1.0f + e);
    return copysignf(t, x);
}

// ---------------- weight packing kernel ----------------
// Packs W (G3 x HDIM, fp32) into bf16 WMMA-B fragments:
// frag[nt][kt][lane][p]; lane: n = nt*16 + (lane&15), kh = (lane>=16)*8;
// p<8 -> K = kt*32 + kh + p ; p>=8 -> K = kt*32 + 16 + kh + (p-8).
__global__ void pack_weights(const float* __restrict__ w0, const float* __restrict__ w1,
                             const float* __restrict__ w2, const float* __restrict__ w3,
                             bf16* __restrict__ out) {
    int gid = blockIdx.x * blockDim.x + threadIdx.x;
    const int RECS = (G3 / 16) * KT * 32;  // 49152 lane-records per matrix
    int m = gid / RECS;
    if (m >= 4) return;
    int rec = gid - m * RECS;
    const float* W = (m == 0) ? w0 : (m == 1) ? w1 : (m == 2) ? w2 : w3;
    int nt = rec / (KT * 32);
    int r2 = rec - nt * (KT * 32);
    int kt = r2 >> 5;
    int L  = r2 & 31;
    int n  = nt * 16 + (L & 15);
    int kh = (L >> 4) << 3;
    bf16* o = out + ((size_t)m * RECS + rec) * 16;
#pragma unroll
    for (int p = 0; p < 16; ++p) {
        int ko = kt * 32 + ((p < 8) ? (kh + p) : (16 + kh + (p - 8)));
        o[p] = (bf16)W[(size_t)n * HDIM + ko];
    }
}

// ---------------- fused GRU step ----------------
// MODE 0: encoder layer0 (input from global x, K_in=4, VALU side-gemm)
// MODE 1: encoder layer1 (input from LDS a2 via WMMA)
// MODE 2: decoder       (input from predbuf, K_in=2, VALU side-gemm)
template <int MODE>
__device__ __forceinline__ void gru_step(
    const bf16* __restrict__ hcur, bf16* __restrict__ hnxt,
    const bf16* __restrict__ a2,
    const float* __restrict__ xrow, int xrstride,
    const float* __restrict__ smallW,
    const float* __restrict__ predbuf,
    const bf16* __restrict__ whhF, const bf16* __restrict__ wihF,
    const float* __restrict__ bih, const float* __restrict__ bhh,
    int lane, int wv) {
    for (int i = 0; i < UNITS_PW; ++i) {
        int unit = i * WAVES + wv;
        int mp  = unit >> 5;          // m-pair: 0..1
        int jg  = unit & 31;          // column group
        int mt0 = mp * 2, mt1 = mp * 2 + 1;
        f32x8 ar0 = (f32x8)0.0f, az0 = (f32x8)0.0f, ahn0 = (f32x8)0.0f, ain0 = (f32x8)0.0f;
        f32x8 ar1 = (f32x8)0.0f, az1 = (f32x8)0.0f, ahn1 = (f32x8)0.0f, ain1 = (f32x8)0.0f;
        // recurrent side: h @ W_hh^T for gates r,z,n — each B fragment feeds 2 M-tiles
#pragma unroll
        for (int kt = 0; kt < KT; ++kt) {
            bf16x16 a0 = ldsA(hcur, mt0, kt, lane);
            bf16x16 a1 = ldsA(hcur, mt1, kt, lane);
            bf16x16 br = ldB(whhF, jg,      kt, lane);
            bf16x16 bz = ldB(whhF, 32 + jg, kt, lane);
            bf16x16 bn = ldB(whhF, 64 + jg, kt, lane);
            ar0  = wmma_bf16(a0, br, ar0);   ar1  = wmma_bf16(a1, br, ar1);
            az0  = wmma_bf16(a0, bz, az0);   az1  = wmma_bf16(a1, bz, az1);
            ahn0 = wmma_bf16(a0, bn, ahn0);  ahn1 = wmma_bf16(a1, bn, ahn1);
        }
        if constexpr (MODE == 1) {  // input side: a2 @ W_ih^T (r,z merged; n separate)
#pragma unroll
            for (int kt = 0; kt < KT; ++kt) {
                bf16x16 a0 = ldsA(a2, mt0, kt, lane);
                bf16x16 a1 = ldsA(a2, mt1, kt, lane);
                bf16x16 br = ldB(wihF, jg,      kt, lane);
                bf16x16 bz = ldB(wihF, 32 + jg, kt, lane);
                bf16x16 bn = ldB(wihF, 64 + jg, kt, lane);
                ar0  = wmma_bf16(a0, br, ar0);   ar1  = wmma_bf16(a1, br, ar1);
                az0  = wmma_bf16(a0, bz, az0);   az1  = wmma_bf16(a1, bz, az1);
                ain0 = wmma_bf16(a0, bn, ain0);  ain1 = wmma_bf16(a1, bn, ain1);
            }
        }
        int j = jg * 16 + (lane & 15);
        float bir = bih[j],            bhr = bhh[j];
        float biz = bih[HDIM + j],     bhz = bhh[HDIM + j];
        float bin = bih[2 * HDIM + j], bhn = bhh[2 * HDIM + j];
        constexpr int KIN = (MODE == 0) ? 4 : 2;
        float wir[KIN], wiz[KIN], win_[KIN];
        if constexpr (MODE != 1) {
#pragma unroll
            for (int k = 0; k < KIN; ++k) {
                wir[k]  = smallW[(size_t)j * KIN + k];
                wiz[k]  = smallW[(size_t)(HDIM + j) * KIN + k];
                win_[k] = smallW[(size_t)(2 * HDIM + j) * KIN + k];
            }
        }
#pragma unroll
        for (int m = 0; m < 2; ++m) {
            int mt = mp * 2 + m;
            const f32x8& ar  = m ? ar1  : ar0;
            const f32x8& az  = m ? az1  : az0;
            const f32x8& ahn = m ? ahn1 : ahn0;
            const f32x8& ain = m ? ain1 : ain0;
#pragma unroll
            for (int v = 0; v < 8; ++v) {
                int row = mt * 16 + v + ((lane >> 4) << 3);  // C-matrix layout
                float ir = bir, iz = biz, inn = bin;
                if constexpr (MODE == 0) {
                    const float* xr = xrow + (size_t)row * xrstride;
#pragma unroll
                    for (int k = 0; k < 4; ++k) {
                        float xv = xr[k];
                        ir += xv * wir[k]; iz += xv * wiz[k]; inn += xv * win_[k];
                    }
                } else if constexpr (MODE == 2) {
                    float x0 = predbuf[row * 2 + 0], x1 = predbuf[row * 2 + 1];
                    ir  += x0 * wir[0]  + x1 * wir[1];
                    iz  += x0 * wiz[0]  + x1 * wiz[1];
                    inn += x0 * win_[0] + x1 * win_[1];
                } else {
                    inn += ain[v];
                }
                float r = sigmoidf_(ar[v] + ir + bhr);
                float z = sigmoidf_(az[v] + iz + bhz);
                float n = tanhf_(inn + r * (ahn[v] + bhn));
                float hold = (float)hcur[row * LDS_STRIDE + j];
                hnxt[row * LDS_STRIDE + j] = (bf16)((1.0f - z) * n + z * hold);
            }
        }
    }
    __syncthreads();
}

// ---------------- main fused network kernel ----------------
__global__ void __launch_bounds__(TPB, 1) gru_net(
    const float* __restrict__ x,
    const float* __restrict__ Wih0, const float* __restrict__ bih0, const float* __restrict__ bhh0,
    const float* __restrict__ bih1, const float* __restrict__ bhh1,
    const float* __restrict__ Wihd, const float* __restrict__ bihd, const float* __restrict__ bhhd,
    const float* __restrict__ Wfc,  const float* __restrict__ bfc,
    const bf16* __restrict__ frags, float* __restrict__ out) {
    extern __shared__ char smemraw[];
    bf16* h0a = (bf16*)smemraw;
    bf16* h0b = h0a + TILE_B * LDS_STRIDE;
    bf16* h1a = h0b + TILE_B * LDS_STRIDE;
    bf16* h1b = h1a + TILE_B * LDS_STRIDE;
    float* predbuf = (float*)(h1b + TILE_B * LDS_STRIDE);

    const int tid = threadIdx.x, lane = tid & 31, wv = tid >> 5;
    const int b0 = blockIdx.x * TILE_B;

    const bf16* Whh0F = frags + (size_t)0 * FRAG_MAT;
    const bf16* Wih1F = frags + (size_t)1 * FRAG_MAT;
    const bf16* Whh1F = frags + (size_t)2 * FRAG_MAT;
    const bf16* WhhdF = frags + (size_t)3 * FRAG_MAT;

    for (int idx = tid; idx < TILE_B * LDS_STRIDE; idx += TPB) {
        h0a[idx] = (bf16)0.0f;
        h1a[idx] = (bf16)0.0f;
    }
    __syncthreads();

    bf16 *h0c = h0a, *h0n = h0b, *h1c = h1a, *h1n = h1b;

    // -------- encoder: layer0 -> layer1 fused per step --------
    for (int t = 0; t < T_OBS; ++t) {
        const float* xrow = x + ((size_t)b0 * T_OBS + t) * IN_DIM;
        gru_step<0>(h0c, h0n, nullptr, xrow, T_OBS * IN_DIM, Wih0, nullptr,
                    Whh0F, nullptr, bih0, bhh0, lane, wv);
        { bf16* tmp = h0c; h0c = h0n; h0n = tmp; }
        gru_step<1>(h1c, h1n, h0c, nullptr, 0, nullptr, nullptr,
                    Whh1F, Wih1F, bih1, bhh1, lane, wv);
        { bf16* tmp = h1c; h1c = h1n; h1n = tmp; }
    }

    // -------- decoder init input = x[:, -1, :2] --------
    if (tid < TILE_B * OUT_DIM) {
        int row = tid >> 1, o = tid & 1;
        predbuf[tid] = x[((size_t)(b0 + row) * T_OBS + (T_OBS - 1)) * IN_DIM + o];
    }
    __syncthreads();

    // -------- decoder + FC head --------
    for (int t = 0; t < T_PRED; ++t) {
        gru_step<2>(h1c, h1n, nullptr, nullptr, 0, Wihd, predbuf,
                    WhhdF, nullptr, bihd, bhhd, lane, wv);
        { bf16* tmp = h1c; h1c = h1n; h1n = tmp; }  // h1c = new hidden
        // pred = h_new @ W_fc^T + b_fc  (per-wave rows, lane-strided dot + shuffle reduce)
        for (int rr = 0; rr < TILE_B / WAVES; ++rr) {
            int row = wv * (TILE_B / WAVES) + rr;
            float s0 = 0.0f, s1 = 0.0f;
            for (int c = lane; c < HDIM; c += 32) {
                float hv = (float)h1c[row * LDS_STRIDE + c];
                s0 += hv * Wfc[c];
                s1 += hv * Wfc[HDIM + c];
            }
#pragma unroll
            for (int m = 16; m >= 1; m >>= 1) {
                s0 += __shfl_xor(s0, m, 32);
                s1 += __shfl_xor(s1, m, 32);
            }
            if (lane == 0) {
                s0 += bfc[0];
                s1 += bfc[1];
                size_t ob = ((size_t)(b0 + row) * T_PRED + t) * OUT_DIM;
                out[ob + 0] = s0;
                out[ob + 1] = s1;
                predbuf[row * 2 + 0] = s0;
                predbuf[row * 2 + 1] = s1;
            }
        }
        __syncthreads();
    }
}

// ---------------- host entry ----------------
extern "C" void kernel_launch(void* const* d_in, const int* in_sizes, int n_in,
                              void* d_out, int out_size, void* d_ws, size_t ws_size,
                              hipStream_t stream) {
    (void)in_sizes; (void)n_in; (void)out_size; (void)ws_size;
    const float* x    = (const float*)d_in[0];
    const float* Wih0 = (const float*)d_in[1];
    const float* Whh0 = (const float*)d_in[2];
    const float* bih0 = (const float*)d_in[3];
    const float* bhh0 = (const float*)d_in[4];
    const float* Wih1 = (const float*)d_in[5];
    const float* Whh1 = (const float*)d_in[6];
    const float* bih1 = (const float*)d_in[7];
    const float* bhh1 = (const float*)d_in[8];
    const float* Wihd = (const float*)d_in[9];
    const float* Whhd = (const float*)d_in[10];
    const float* bihd = (const float*)d_in[11];
    const float* bhhd = (const float*)d_in[12];
    const float* Wfc  = (const float*)d_in[13];
    const float* bfc  = (const float*)d_in[14];
    float* out  = (float*)d_out;
    bf16* frags = (bf16*)d_ws;   // needs 4 * 786432 * 2 B = 6 MB

    // one-time per call: fp32 -> bf16 fragment packing of the 4 big (1536x512) matrices
    const int RECS_TOTAL = 4 * (G3 / 16) * KT * 32;  // 196608
    pack_weights<<<(RECS_TOTAL + 255) / 256, 256, 0, stream>>>(Whh0, Wih1, Whh1, Whhd, frags);

    hipFuncSetAttribute(reinterpret_cast<const void*>(gru_net),
                        hipFuncAttributeMaxDynamicSharedMemorySize, SMEM_BYTES);
    gru_net<<<BATCH / TILE_B, TPB, SMEM_BYTES, stream>>>(
        x, Wih0, bih0, bhh0, bih1, bhh1, Wihd, bihd, bhhd, Wfc, bfc, frags, out);
}
